// GroupedExperts_609885356950
// MI455X (gfx1250) — matrix operand order repaired and checked by reading, not compiled, not run on hardware
//
#include <hip/hip_runtime.h>
#include <hip/hip_bf16.h>

#define N_EXPERTS 8
#define D_MODEL 512
#define HIDDEN 1024
#define TOKENS_PER_EXPERT 1024

#define BM 128
#define BK 32
#define BN1 64      // h-columns per block in fused gemm1 (h1 and h2 tiles both loaded)
#define BN2 128     // out-columns per block in gemm2
#define LDSR 40     // padded LDS row stride in bf16 elements (80B: 16B-aligned rows, bank-spread)

typedef __bf16 v16bf __attribute__((ext_vector_type(16)));
typedef __bf16 v2bf  __attribute__((ext_vector_type(2)));
typedef float  v8f   __attribute__((ext_vector_type(8)));

// Pairwise f32->bf16 through the native __bf16 type: on gfx1250 fptrunc(f32->bf16)
// is legal and a pair selects v_cvt_pk_bf16_f32 (RNE), ~1 VALU per 2 elements.
static __device__ __forceinline__ unsigned int pk2bf(float a, float b) {
    union { v2bf v; unsigned int u; } r;
    r.v = v2bf{(__bf16)a, (__bf16)b};
    return r.u;
}
static __device__ __forceinline__ uint2 pk4bf(float4 v) {
    return make_uint2(pk2bf(v.x, v.y), pk2bf(v.z, v.w));
}

// ---------------- Kernel 1: h = silu(X @ W1^T) * (X @ W2^T), stored bf16 ----------------
__global__ __launch_bounds__(256)
void k_gemm1_swiglu(const float* __restrict__ X, const float* __restrict__ W12,
                    const int* __restrict__ starts, __bf16* __restrict__ H) {
    __shared__ unsigned short sA[2][BM * LDSR];
    __shared__ unsigned short sB1[2][BN1 * LDSR];
    __shared__ unsigned short sB2[2][BN1 * LDSR];

    const int tid  = threadIdx.x;
    const int e    = blockIdx.z;
    const int row0 = starts[e] + blockIdx.x * BM;
    const int n0   = blockIdx.y * BN1;
    const float* We = W12 + (size_t)e * (2 * HIDDEN) * D_MODEL;

    const int lane = tid & 31, wave = tid >> 5;
    const int wm = wave & 3, wn = wave >> 2;        // wave -> 32-row group x 32-col group
    const int half = lane >> 4, lm = lane & 15;
    const int kloA = half * 16, kloB = half * 16;

    v8f acc1[2][2], acc2[2][2];
#pragma unroll
    for (int mi = 0; mi < 2; ++mi)
#pragma unroll
        for (int ni = 0; ni < 2; ++ni) { acc1[mi][ni] = v8f{}; acc2[mi][ni] = v8f{}; }

    const int r8 = tid >> 3, c4 = tid & 7;          // loader: 8 threads/row, float4 chunks

    auto load_tile = [&](int k0, int buf) {
#pragma unroll
        for (int i = 0; i < 4; ++i) {               // A: 128 rows x 32 f32 -> bf16
            const int r = r8 + i * 32;
            float4 v = *(const float4*)(X + (size_t)(row0 + r) * D_MODEL + k0 + c4 * 4);
            *(uint2*)&sA[buf][r * LDSR + c4 * 4] = pk4bf(v);
        }
#pragma unroll
        for (int i = 0; i < 2; ++i) {               // B1 (h1 rows), B2 (h2 rows)
            const int r = r8 + i * 32;
            float4 v1 = *(const float4*)(We + (size_t)(n0 + r) * D_MODEL + k0 + c4 * 4);
            *(uint2*)&sB1[buf][r * LDSR + c4 * 4] = pk4bf(v1);
            float4 v2 = *(const float4*)(We + (size_t)(HIDDEN + n0 + r) * D_MODEL + k0 + c4 * 4);
            *(uint2*)&sB2[buf][r * LDSR + c4 * 4] = pk4bf(v2);
        }
    };

    constexpr int NSTEP = D_MODEL / BK;             // 16
    load_tile(0, 0);
    int cur = 0;
    for (int ks = 0; ks < NSTEP; ++ks) {
        __syncthreads();                            // tile[cur] visible; tile[cur^1] free
        if (ks + 1 < NSTEP) load_tile((ks + 1) * BK, cur ^ 1);

        v16bf a[2], b1[2], b2[2];
#pragma unroll
        for (int mi = 0; mi < 2; ++mi) {            // A frag: 16 contiguous K at kloA
            const unsigned short* p = &sA[cur][(wm * 32 + mi * 16 + lm) * LDSR + kloA];
            ((uint4*)&a[mi])[0] = *(const uint4*)(p);
            ((uint4*)&a[mi])[1] = *(const uint4*)(p + 8);
        }
#pragma unroll
        for (int ni = 0; ni < 2; ++ni) {            // B frag: 16 contiguous K at kloB
            const unsigned short* p1 = &sB1[cur][(wn * 32 + ni * 16 + lm) * LDSR + kloB];
            ((uint4*)&b1[ni])[0] = *(const uint4*)(p1);
            ((uint4*)&b1[ni])[1] = *(const uint4*)(p1 + 8);
            const unsigned short* p2 = &sB2[cur][(wn * 32 + ni * 16 + lm) * LDSR + kloB];
            ((uint4*)&b2[ni])[0] = *(const uint4*)(p2);
            ((uint4*)&b2[ni])[1] = *(const uint4*)(p2 + 8);
        }
#pragma unroll
        for (int mi = 0; mi < 2; ++mi)
#pragma unroll
            for (int ni = 0; ni < 2; ++ni) {
                acc1[mi][ni] = __builtin_amdgcn_wmma_f32_16x16x32_bf16(
                    false, a[mi], false, b1[ni], (short)0, acc1[mi][ni], false, false);
                acc2[mi][ni] = __builtin_amdgcn_wmma_f32_16x16x32_bf16(
                    false, a[mi], false, b2[ni], (short)0, acc2[mi][ni], false, false);
            }
        cur ^= 1;
    }

#pragma unroll
    for (int mi = 0; mi < 2; ++mi)
#pragma unroll
        for (int ni = 0; ni < 2; ++ni) {
            const int c = n0 + wn * 32 + ni * 16 + lm;
#pragma unroll
            for (int v = 0; v < 8; ++v) {           // C/D layout: m = v + 8*half, n = lane%16
                const int t = row0 + wm * 32 + mi * 16 + half * 8 + v;
                float x1 = acc1[mi][ni][v], x2 = acc2[mi][ni][v];
                // silu(x1)*x2 with fast reciprocal (v_rcp_f32), not IEEE division
                float h = x1 * __builtin_amdgcn_rcpf(1.0f + __expf(-x1)) * x2;
                H[(size_t)t * HIDDEN + c] = (__bf16)h;
            }
        }
}

// ---------------- Kernel 2: out = H @ W3^T (f32 out) ----------------
__global__ __launch_bounds__(256)
void k_gemm2(const __bf16* __restrict__ H, const float* __restrict__ W3,
             const int* __restrict__ starts, float* __restrict__ Out) {
    __shared__ unsigned short sA[2][BM * LDSR];
    __shared__ unsigned short sB[2][BN2 * LDSR];

    const int tid  = threadIdx.x;
    const int e    = blockIdx.z;
    const int row0 = starts[e] + blockIdx.x * BM;
    const int n0   = blockIdx.y * BN2;
    const float* We = W3 + (size_t)e * D_MODEL * HIDDEN;

    const int lane = tid & 31, wave = tid >> 5;
    const int wm = wave & 3, wn = wave >> 2;        // wave -> 32 rows x 64 cols
    const int half = lane >> 4, lm = lane & 15;
    const int kloA = half * 16, kloB = half * 16;

    v8f acc[2][4];
#pragma unroll
    for (int mi = 0; mi < 2; ++mi)
#pragma unroll
        for (int ni = 0; ni < 4; ++ni) acc[mi][ni] = v8f{};

    const int r4 = tid >> 2, c8 = tid & 3;          // A loader: 4 x 16B chunks per row
    const int r8 = tid >> 3, c4 = tid & 7;          // B loader: 8 x float4 per row

    auto load_tile = [&](int k0, int buf) {
#pragma unroll
        for (int i = 0; i < 2; ++i) {
            // A: raw bf16 copy global->LDS via CDNA5 async-to-LDS DMA (GVS mode):
            // vdst = per-lane LDS byte offset (low 32 bits of flat LDS addr),
            // voffset = per-lane global byte offset, saddr = H base (SGPR pair).
            const int r = r4 + i * 64;
            unsigned lds_off = (unsigned)(size_t)&sA[buf][r * LDSR + c8 * 8];
            unsigned goff = (unsigned)(((size_t)(row0 + r) * HIDDEN + k0 + c8 * 8) * sizeof(__bf16));
            asm volatile("global_load_async_to_lds_b128 %0, %1, %2"
                         :: "v"(lds_off), "v"(goff), "s"(H) : "memory");
        }
#pragma unroll
        for (int i = 0; i < 4; ++i) {               // B: w3 f32 -> bf16
            const int r = r8 + i * 32;
            float4 v = *(const float4*)(We + (size_t)(n0 + r) * HIDDEN + k0 + c4 * 4);
            *(uint2*)&sB[buf][r * LDSR + c4 * 4] = pk4bf(v);
        }
    };

    constexpr int NSTEP = HIDDEN / BK;              // 32
    load_tile(0, 0);
    int cur = 0;
    for (int ks = 0; ks < NSTEP; ++ks) {
        // async LDS writes are tracked by ASYNCcnt, which __syncthreads does not wait on
        asm volatile("s_wait_asynccnt 0" ::: "memory");
        __syncthreads();
        if (ks + 1 < NSTEP) load_tile((ks + 1) * BK, cur ^ 1);

        v16bf a[2], b[4];
#pragma unroll
        for (int mi = 0; mi < 2; ++mi) {
            const unsigned short* p = &sA[cur][(wm * 32 + mi * 16 + lm) * LDSR + kloA];
            ((uint4*)&a[mi])[0] = *(const uint4*)(p);
            ((uint4*)&a[mi])[1] = *(const uint4*)(p + 8);
        }
#pragma unroll
        for (int ni = 0; ni < 4; ++ni) {
            const unsigned short* p = &sB[cur][(wn * 64 + ni * 16 + lm) * LDSR + kloB];
            ((uint4*)&b[ni])[0] = *(const uint4*)(p);
            ((uint4*)&b[ni])[1] = *(const uint4*)(p + 8);
        }
#pragma unroll
        for (int mi = 0; mi < 2; ++mi)
#pragma unroll
            for (int ni = 0; ni < 4; ++ni)
                acc[mi][ni] = __builtin_amdgcn_wmma_f32_16x16x32_bf16(
                    false, a[mi], false, b[ni], (short)0, acc[mi][ni], false, false);
        cur ^= 1;
    }

#pragma unroll
    for (int mi = 0; mi < 2; ++mi)
#pragma unroll
        for (int ni = 0; ni < 4; ++ni) {
            const int d = n0 + wn * 64 + ni * 16 + lm;
#pragma unroll
            for (int v = 0; v < 8; ++v) {
                const int t = row0 + wm * 32 + mi * 16 + half * 8 + v;
                Out[(size_t)t * D_MODEL + d] = acc[mi][ni][v];
            }
        }
}

extern "C" void kernel_launch(void* const* d_in, const int* in_sizes, int n_in,
                              void* d_out, int out_size, void* d_ws, size_t ws_size,
                              hipStream_t stream) {
    const float* X      = (const float*)d_in[0];   // (8192, 512)
    const float* W12    = (const float*)d_in[1];   // (8, 2048, 512)
    const float* W3     = (const float*)d_in[2];   // (8, 512, 1024)
    const int*   starts = (const int*)d_in[3];     // (8,)
    float*       Out    = (float*)d_out;           // (8192, 512)
    __bf16*      H      = (__bf16*)d_ws;           // (8192, 1024) bf16 intermediate (16.8 MB)

    dim3 g1(TOKENS_PER_EXPERT / BM, HIDDEN / BN1, N_EXPERTS);   // (8, 16, 8)
    k_gemm1_swiglu<<<g1, 256, 0, stream>>>(X, W12, starts, H);

    dim3 g2(TOKENS_PER_EXPERT / BM, D_MODEL / BN2, N_EXPERTS);  // (8, 4, 8)
    k_gemm2<<<g2, 256, 0, stream>>>(H, W3, starts, Out);
}
